// DownSample_73967926771953
// MI455X (gfx1250) — matrix operand order repaired and checked by reading, not compiled, not run on hardware
//
#include <hip/hip_runtime.h>
#include <hip/hip_bf16.h>
#include <math.h>

#define BB 16
#define CIN 64
#define NN 4096
#define OO 128
#define MM 1024
#define CC 67
#define KPAD 96
#define BM (BB*MM)

// 1/sqrt(128) * log2(e): logits produced directly in log2 domain for exp2-softmax
static constexpr float QSCALE = 0.08838834764831845f * 1.4426950408889634f;

typedef _Float16 h16;
typedef __attribute__((ext_vector_type(16))) _Float16 v16h;
typedef __attribute__((ext_vector_type(8)))  float    v8f;
typedef __attribute__((ext_vector_type(4)))  unsigned int u32x4;
typedef __attribute__((ext_vector_type(4)))  int i32x4;
typedef __attribute__((ext_vector_type(8)))  int i32x8;

union Frag16 { v16h v; u32x4 u4[2]; };

// LDS strides (halfs) chosen for conflict-free ds_read_b128 fragment loads
#define KROW 136   // 128 + 8 pad  (272B row: bank stride 4)
#define VROW 40    // 32 + 8 pad   (80B row:  bank stride 20)
#define PROW 40    // 32 + 8 pad

#if defined(__has_builtin)
#  if __has_builtin(__builtin_amdgcn_tensor_load_to_lds)
#    define HAVE_TDM 1
#  endif
#  if __has_builtin(__builtin_amdgcn_permlane16)
#    define HAVE_PERMLANE 1
#  endif
#endif
#ifndef HAVE_TDM
#  define HAVE_TDM 0
#endif
#ifndef HAVE_PERMLANE
#  define HAVE_PERMLANE 0
#endif

__device__ __forceinline__ v8f wmma_f16(v16h a, v16h b, v8f c) {
  // D = A(16x32 f16) * B(32x16 f16) + C(16x16 f32)
  return __builtin_amdgcn_wmma_f32_16x16x32_f16(false, a, false, b, (short)0, c, false, false);
}

// ---- 16-lane-row butterfly reductions on the VALU (v_permlane16_b32) --------
// Selectors encode lane -> lane^m within each 16-lane row; wave32 has two rows
// (lanes 0-15 / 16-31) reduced independently -- exactly the softmax row halves.
#if HAVE_PERMLANE
__device__ __forceinline__ float pl16(float x, unsigned s0, unsigned s1) {
  return __uint_as_float(
      __builtin_amdgcn_permlane16(__float_as_uint(x), __float_as_uint(x), s0, s1, false, false));
}
__device__ __forceinline__ float red16_max(float t) {
  t = fmaxf(t, pl16(t, 0x67452301u, 0xefcdab89u));   // xor 1
  t = fmaxf(t, pl16(t, 0x54761032u, 0xdcfe98bau));   // xor 2
  t = fmaxf(t, pl16(t, 0x32107654u, 0xba98fedcu));   // xor 4
  t = fmaxf(t, pl16(t, 0xfedcba98u, 0x76543210u));   // xor 8
  return t;
}
__device__ __forceinline__ float red16_sum(float t) {
  t += pl16(t, 0x67452301u, 0xefcdab89u);
  t += pl16(t, 0x54761032u, 0xdcfe98bau);
  t += pl16(t, 0x32107654u, 0xba98fedcu);
  t += pl16(t, 0xfedcba98u, 0x76543210u);
  return t;
}
#else
__device__ __forceinline__ float red16_max(float t) {
  #pragma unroll
  for (int off = 8; off > 0; off >>= 1) t = fmaxf(t, __shfl_xor(t, off, 32));
  return t;
}
__device__ __forceinline__ float red16_sum(float t) {
  #pragma unroll
  for (int off = 8; off > 0; off >>= 1) t += __shfl_xor(t, off, 32);
  return t;
}
#endif

#if HAVE_TDM
// Build a TDM descriptor (D# groups per CDNA5 ISA ch.8) and issue tensor_load_to_lds.
// 2D tile: tile0 (X, elements) x tile1 (Y rows, 0 = 1D), row stride stride0 (elements),
// 2-byte elements, LDS padding: 2^(padint+3) bytes interval, (padamt+1) dwords pad.
// This toolchain exposes the 6-arg builtin:
//   (u32x4 g0, i32x8 g1, i32x4 g2, i32x4 g3, i32x8 g4, i32 cpol)
__device__ __forceinline__ void tdm_load(unsigned lds_addr, const h16* gptr,
                                         unsigned dim0, unsigned dim1,
                                         unsigned tile0, unsigned tile1,
                                         unsigned stride0,
                                         unsigned padint, unsigned padamt) {
  unsigned long long ga = (unsigned long long)(size_t)gptr;
  u32x4 g0;
  g0.x = 1u;                                        // count=1 (valid, user mode)
  g0.y = lds_addr;                                  // LDS byte address
  g0.z = (unsigned)ga;                              // global_addr[31:0]
  g0.w = (unsigned)(ga >> 32) | (2u << 30);         // global_addr[56:32] | type=2
  i32x8 g1;
  g1[0] = (int)((1u << 16) | (1u << 20) | (padint << 22) | (padamt << 25)); // 2B elems, pad_enable
  g1[1] = (int)((dim0 & 0xffffu) << 16);                                    // tensor_dim0[15:0]
  g1[2] = (int)(((dim0 >> 16) & 0xffffu) | ((dim1 & 0xffffu) << 16));       // dim0[31:16] | dim1[15:0]
  g1[3] = (int)(((dim1 >> 16) & 0xffffu) | ((tile0 & 0xffffu) << 16));      // dim1[31:16] | tile_dim0
  g1[4] = (int)(tile1 & 0xffffu);                                           // tile_dim1 (tile_dim2=0)
  g1[5] = (int)stride0;                                                     // dim0_stride[31:0]
  g1[6] = 0;                                                                // dim0_stride[47:32] | dim1_stride[15:0]
  g1[7] = 0;
  i32x4 z4 = {0, 0, 0, 0};
  i32x8 z8 = {0, 0, 0, 0, 0, 0, 0, 0};
  __builtin_amdgcn_tensor_load_to_lds(g0, g1, z4, z4, z8, 0);
}
#endif

// ---------------- weight convert (f32 -> f16, K-padded, optional scale) ------
__global__ void k_convw(const float* __restrict__ src, h16* __restrict__ dst,
                        int kin, int kpad, float scale) {
  int g = blockIdx.x * blockDim.x + threadIdx.x;
  int r = g / kpad, c = g % kpad;
  dst[g] = (c < kin) ? (h16)(src[r * kin + c] * scale) : (h16)0.f;
}

// ---------------- concat x & xt -> xcT [B, N, 96] f16 (zero padded) ----------
__global__ void k_concat(const float* __restrict__ x, const float* __restrict__ xt,
                         h16* __restrict__ xcT) {
  int g = blockIdx.x * blockDim.x + threadIdx.x; // over B*N
  int b = g >> 12, n = g & (NN - 1);
  h16* trow = xcT + (size_t)g * KPAD;
  #pragma unroll 4
  for (int c = 0; c < CIN; ++c) {
    float v = x[((size_t)b * CIN + c) * NN + n];
    trow[c] = (h16)v;
  }
  #pragma unroll
  for (int c = 0; c < 3; ++c) {
    float v = xt[((size_t)b * 3 + c) * NN + n];
    trow[CIN + c] = (h16)v;
  }
  #pragma unroll
  for (int c = CC; c < KPAD; ++c) trow[c] = (h16)0.f;
}

// ---------------- farthest point sampling (serial over M) --------------------
__global__ __launch_bounds__(1024) void k_fps(const h16* __restrict__ xcT,
                                              int* __restrict__ idx_out) {
  int b = blockIdx.x;
  int tid = threadIdx.x;
  int lane = tid & 31, wv = tid >> 5;
  __shared__ float s_cent[KPAD];
  __shared__ float s_wval[32];
  __shared__ int   s_widx[32];
  __shared__ int   s_far;
  float dmin[4];
  #pragma unroll
  for (int p = 0; p < 4; ++p) dmin[p] = 1e10f;
  int far = 0;
  const h16* base = xcT + (size_t)b * NN * KPAD;
  for (int it = 0; it < MM; ++it) {
    if (tid == 0) idx_out[b * MM + it] = far;  // emit BEFORE update (matches ref)
    if (tid < KPAD) s_cent[tid] = (float)base[(size_t)far * KPAD + tid];
    __syncthreads();
    float bestv = -1.f; int besti = 0;
    #pragma unroll
    for (int p = 0; p < 4; ++p) {
      int n = p * 1024 + tid;
      const u32x4* src = (const u32x4*)(base + (size_t)n * KPAD);
      union { u32x4 u[12]; h16 h[96]; } pt;
      #pragma unroll
      for (int j = 0; j < 12; ++j) pt.u[j] = src[j];
      float d = 0.f;
      #pragma unroll
      for (int c = 0; c < KPAD; ++c) {    // padded dims contribute 0
        float df = (float)pt.h[c] - s_cent[c];
        d = fmaf(df, df, d);
      }
      dmin[p] = fminf(dmin[p], d);
      if (dmin[p] > bestv) { bestv = dmin[p]; besti = n; } // strict >: first max wins
    }
    // wave reduce (argmax, low index on ties)
    #pragma unroll
    for (int off = 16; off > 0; off >>= 1) {
      float ov = __shfl_xor(bestv, off, 32);
      int   oi = __shfl_xor(besti, off, 32);
      if (ov > bestv || (ov == bestv && oi < besti)) { bestv = ov; besti = oi; }
    }
    if (lane == 0) { s_wval[wv] = bestv; s_widx[wv] = besti; }
    __syncthreads();
    if (wv == 0) {
      bestv = s_wval[lane]; besti = s_widx[lane];
      #pragma unroll
      for (int off = 16; off > 0; off >>= 1) {
        float ov = __shfl_xor(bestv, off, 32);
        int   oi = __shfl_xor(besti, off, 32);
        if (ov > bestv || (ov == bestv && oi < besti)) { bestv = ov; besti = oi; }
      }
      if (lane == 0) s_far = besti;
    }
    __syncthreads();
    far = s_far;
  }
}

// ---------------- gather centroids: centT [B, M, 96] f16 ---------------------
__global__ void k_gather(const h16* __restrict__ xcT, const int* __restrict__ idx,
                         h16* __restrict__ centT) {
  size_t g = (size_t)blockIdx.x * blockDim.x + threadIdx.x; // B*M*96
  int c = (int)(g % KPAD);
  size_t bm = g / KPAD;
  int m = (int)(bm % MM), b = (int)(bm / MM);
  int n = idx[b * MM + m];
  centT[g] = xcT[((size_t)b * NN + n) * KPAD + c];
}

// ---------------- generic WMMA GEMM: D = W[128 x KP] * X[b] ------------------
// X stored column-major per column: X[b][col][k] (K contiguous).
// OUT_MODE 0: Yt f16 [B, ncols, 128]; 1: Y f16 [B, 128, ncols]; 2: Yt f32.
template<int KP, int OUT_MODE, bool BIAS>
__global__ __launch_bounds__(128) void k_gemm(const h16* __restrict__ W,
                                              const h16* __restrict__ X,
                                              const float* __restrict__ bias,
                                              void* __restrict__ Yv, int ncols) {
  int lane = threadIdx.x & 31;
  int wv = threadIdx.x >> 5;
  int l15 = lane & 15, hf = lane >> 4;
  int colb = blockIdx.x * 16;
  int eb = (blockIdx.y * 4 + wv) * 16;
  int b = blockIdx.z;
  const h16* wrow = W + (size_t)(eb + l15) * KP + (hf ? 8 : 0);
  const h16* xcol = X + ((size_t)b * ncols + colb + l15) * KP + (hf ? 8 : 0);
  v8f acc = {};
  #pragma unroll
  for (int kc = 0; kc < KP / 32; ++kc) {
    Frag16 a, bf;
    a.u4[0]  = *(const u32x4*)(wrow + kc * 32);
    a.u4[1]  = *(const u32x4*)(wrow + kc * 32 + 16);
    bf.u4[0] = *(const u32x4*)(xcol + kc * 32);
    bf.u4[1] = *(const u32x4*)(xcol + kc * 32 + 16);
    acc = wmma_f16(a.v, bf.v, acc);
  }
  #pragma unroll
  for (int r = 0; r < 8; ++r) {
    int e = eb + r + (hf ? 8 : 0);
    int col = colb + l15;
    float val = acc[r];
    if (BIAS) val += bias[e];
    if (OUT_MODE == 0)      ((h16*)Yv)[((size_t)b * ncols + col) * OO + e] = (h16)val;
    else if (OUT_MODE == 1) ((h16*)Yv)[((size_t)b * OO + e) * ncols + col] = (h16)val;
    else                    ((float*)Yv)[((size_t)b * ncols + col) * OO + e] = val;
  }
}

// ---------------- flash attention -------------------------------------------
// Block = 4 waves (one 16-row M tile each), shared b. k/v tiles for the current
// 32 columns are staged in LDS: TDM double-buffered (wave 0 issues
// tensor_load_to_lds with row padding for conflict-free ds_read_b128), all
// waves consume. Softmax runs in the exp2 domain (log2e folded into Wq) with
// permlane16 VALU butterfly reductions.
__global__ __launch_bounds__(128) void k_flash(const h16* __restrict__ qT,
                                               const h16* __restrict__ kT,
                                               const h16* __restrict__ vR,
                                               h16* __restrict__ attT) {
  __shared__ __align__(16) h16 sk[2][32 * KROW];   // k tile: [n_local][d] padded
  __shared__ __align__(16) h16 sv[2][OO * VROW];   // v tile: [d][n_local] padded
  __shared__ __align__(16) h16 lds_p[4][16 * PROW];
  int lane = threadIdx.x & 31;
  int wv = threadIdx.x >> 5;
  int l15 = lane & 15, hf = lane >> 4;
  int b = blockIdx.y;
  int mtile = blockIdx.x * 4 + wv;
  h16* P = lds_p[wv];

  // q A-fragments (16 m x 128 d); log2e/sqrt(O) folded into Wq
  Frag16 qa[4];
  const h16* qrow = qT + ((size_t)b * MM + mtile * 16 + l15) * OO + (hf ? 8 : 0);
  #pragma unroll
  for (int kc = 0; kc < 4; ++kc) {
    qa[kc].u4[0] = *(const u32x4*)(qrow + kc * 32);
    qa[kc].u4[1] = *(const u32x4*)(qrow + kc * 32 + 16);
  }
  v8f acc[8] = {};
  float mi[8], li[8];
  #pragma unroll
  for (int r = 0; r < 8; ++r) { mi[r] = -INFINITY; li[r] = 0.f; }

  const h16* kB = kT + (size_t)b * NN * OO;
  const h16* vB = vR + (size_t)b * OO * NN;
  const int NSTEPS = NN / 32;

#if HAVE_TDM
  if (wv == 0) {
    // prologue: stage step 0 into buffer 0
    tdm_load((unsigned)(size_t)&sk[0][0], kB, 32u * OO, 0u, 32u * OO, 0u, 32u * OO, 5u, 3u);
    tdm_load((unsigned)(size_t)&sv[0][0], vB, (unsigned)NN, (unsigned)OO, 32u, (unsigned)OO,
             (unsigned)NN, 3u, 3u);
  }
#endif

  for (int i = 0; i < NSTEPS; ++i) {
    int buf = i & 1;
    __syncthreads();                      // previous step's readers done with buf^1
#if HAVE_TDM
    if (wv == 0) {
      if (i + 1 < NSTEPS) {
        int n1 = (i + 1) * 32;
        tdm_load((unsigned)(size_t)&sk[buf ^ 1][0], kB + (size_t)n1 * OO,
                 32u * OO, 0u, 32u * OO, 0u, 32u * OO, 5u, 3u);
        tdm_load((unsigned)(size_t)&sv[buf ^ 1][0], vB + n1,
                 (unsigned)NN, (unsigned)OO, 32u, (unsigned)OO, (unsigned)NN, 3u, 3u);
        __builtin_amdgcn_s_wait_tensorcnt(2);   // step i's two TDMs retired
      } else {
        __builtin_amdgcn_s_wait_tensorcnt(0);
      }
    }
    const h16* skb = sk[buf];
    const h16* svb = sv[buf];
#else
    // cooperative fallback: all 128 threads copy step i into buffer 0
    {
      int n0 = i * 32;
      int t = threadIdx.x;
      #pragma unroll
      for (int it2 = 0; it2 < 4; ++it2) {          // k: 64 rows x 8 x 16B
        int idx = t + 128 * it2;
        int row = idx >> 3, ch = idx & 7;
        *(u32x4*)(&sk[0][row * KROW + ch * 8]) =
            *(const u32x4*)(kB + (size_t)(n0 + row) * OO + ch * 8);
      }
      #pragma unroll
      for (int it2 = 0; it2 < 4; ++it2) {          // v: 128 rows x 4 x 16B
        int idx = t + 128 * it2;
        int row = idx >> 2, ch = idx & 3;
        *(u32x4*)(&sv[0][row * VROW + ch * 8]) =
            *(const u32x4*)(vB + (size_t)row * NN + n0 + ch * 8);
      }
    }
    const h16* skb = sk[0];
    const h16* svb = sv[0];
#endif
    __syncthreads();                      // tile published to all waves

    // logits fragments: load all 8 first so ds_reads clause & overlap WMMAs
    Frag16 f[8];
    const h16* kr0 = skb + (size_t)l15 * KROW + (hf ? 8 : 0);
    const h16* kr1 = skb + (size_t)(16 + l15) * KROW + (hf ? 8 : 0);
    #pragma unroll
    for (int kc = 0; kc < 4; ++kc) {
      f[kc].u4[0]     = *(const u32x4*)(kr0 + kc * 32);
      f[kc].u4[1]     = *(const u32x4*)(kr0 + kc * 32 + 16);
      f[4 + kc].u4[0] = *(const u32x4*)(kr1 + kc * 32);
      f[4 + kc].u4[1] = *(const u32x4*)(kr1 + kc * 32 + 16);
    }
    v8f c0 = {}, c1 = {};
    #pragma unroll
    for (int kc = 0; kc < 4; ++kc) {
      c0 = wmma_f16(qa[kc].v, f[kc].v, c0);
      c1 = wmma_f16(qa[kc].v, f[4 + kc].v, c1);
    }
    // online softmax: row r maps to m = r + 8*hf; reduce within 16-lane halves
    float alpha[8];
    #pragma unroll
    for (int r = 0; r < 8; ++r) {
      float t = red16_max(fmaxf(c0[r], c1[r]));
      float mn = fmaxf(mi[r], t);
      float a = exp2f(mi[r] - mn);        // v_exp_f32 is exp2 natively
      float p0 = exp2f(c0[r] - mn);
      float p1 = exp2f(c1[r] - mn);
      float s = red16_sum(p0 + p1);
      li[r] = li[r] * a + s;
      mi[r] = mn;
      alpha[r] = a;
      c0[r] = p0; c1[r] = p1;
    }
    // P tile (16 x 32 f16) via LDS -> A fragment
    int row0 = hf * 8;
    #pragma unroll
    for (int r = 0; r < 8; ++r) {
      P[(row0 + r) * PROW + l15]      = (h16)c0[r];
      P[(row0 + r) * PROW + 16 + l15] = (h16)c1[r];
    }
    asm volatile("s_wait_dscnt 0x0" ::: "memory");
    Frag16 pa;
    const h16* prow = P + l15 * PROW + (hf ? 8 : 0);
    pa.u4[0] = *(const u32x4*)(prow);
    pa.u4[1] = *(const u32x4*)(prow + 16);
    // P @ V : load all 8 B-fragments, then 8 WMMAs with accumulator rescale
    Frag16 bv[8];
    #pragma unroll
    for (int od = 0; od < 8; ++od) {
      const h16* vp = svb + (size_t)(od * 16 + l15) * VROW + (hf ? 8 : 0);
      bv[od].u4[0] = *(const u32x4*)(vp);
      bv[od].u4[1] = *(const u32x4*)(vp + 16);
    }
    #pragma unroll
    for (int od = 0; od < 8; ++od) {
      v8f a8 = acc[od];
      #pragma unroll
      for (int r = 0; r < 8; ++r) a8[r] *= alpha[r];
      acc[od] = wmma_f16(pa.v, bv[od].v, a8);
    }
  }
  #pragma unroll
  for (int od = 0; od < 8; ++od) {
    #pragma unroll
    for (int r = 0; r < 8; ++r) {
      float o = acc[od][r] / li[r];
      attT[((size_t)b * MM + mtile * 16 + r + hf * 8) * OO + od * 16 + l15] = (h16)o;
    }
  }
}

// ---------------- BN statistics --------------------------------------------
__global__ void k_zero(float* __restrict__ p, int n) {
  int g = blockIdx.x * blockDim.x + threadIdx.x;
  if (g < n) p[g] = 0.f;
}

__global__ void k_stats(const float* __restrict__ h32, float* __restrict__ sum,
                        float* __restrict__ sumsq) {
  __shared__ float s1[256], s2[256];
  int t = threadIdx.x;
  int d = t & 127, hf = t >> 7;
  int r0 = blockIdx.x * 128;
  float a = 0.f, b = 0.f;
  for (int r = r0 + hf; r < r0 + 128; r += 2) {
    float v = h32[(size_t)r * OO + d];
    a += v; b += v * v;
  }
  s1[t] = a; s2[t] = b;
  __syncthreads();
  if (t < 128) {
    atomicAdd(&sum[d],   s1[t] + s1[t + 128]);
    atomicAdd(&sumsq[d], s2[t] + s2[t + 128]);
  }
}

__global__ void k_bnfin(const float* __restrict__ sum, const float* __restrict__ sumsq,
                        const float* __restrict__ gamma, const float* __restrict__ beta,
                        float* __restrict__ scaleA, float* __restrict__ shiftA) {
  int t = threadIdx.x;
  if (t < 128) {
    float mean = sum[t] * (1.f / (float)BM);
    float var  = sumsq[t] * (1.f / (float)BM) - mean * mean;  // biased, like torch BN
    float inv  = rsqrtf(var + 1e-5f);
    float sc   = gamma[t] * inv;
    scaleA[t] = sc;
    shiftA[t] = beta[t] - mean * sc;
  }
}

// ---------------- fused BN + LeakyReLU + W2 + b2 -> d_out --------------------
__global__ __launch_bounds__(128) void k_out(const h16* __restrict__ W2f,
                                             const float* __restrict__ h32,
                                             const float* __restrict__ scaleA,
                                             const float* __restrict__ shiftA,
                                             const float* __restrict__ b2,
                                             float* __restrict__ out) {
  int lane = threadIdx.x & 31;
  int wv = threadIdx.x >> 5;
  int l15 = lane & 15, hf = lane >> 4;
  int colb = blockIdx.x * 16;
  int eb = (blockIdx.y * 4 + wv) * 16;
  int b = blockIdx.z;
  const h16* wrow = W2f + (size_t)(eb + l15) * OO + (hf ? 16 : 0);
  const float* hcol = h32 + ((size_t)b * MM + colb + l15) * OO;
  v8f acc = {};
  #pragma unroll
  for (int kc = 0; kc < 4; ++kc) {
    int kbase = kc * 32 + (hf ? 16 : 0);
    Frag16 a, bf;
    a.u4[0] = *(const u32x4*)(wrow + kc * 32);
    a.u4[1] = *(const u32x4*)(wrow + kc * 32 + 8);
    #pragma unroll
    for (int j = 0; j < 16; ++j) {
      float y = fmaf(hcol[kbase + j], scaleA[kbase + j], shiftA[kbase + j]);
      y = (y >= 0.f) ? y : 0.2f * y;
      bf.v[j] = (h16)y;
    }
    acc = wmma_f16(a.v, bf.v, acc);
  }
  #pragma unroll
  for (int r = 0; r < 8; ++r) {
    int e = eb + r + (hf ? 8 : 0);
    int col = colb + l15;
    out[((size_t)b * OO + e) * MM + col] = acc[r] + b2[e];
  }
}

// ---------------- host launcher ---------------------------------------------
extern "C" void kernel_launch(void* const* d_in, const int* in_sizes, int n_in,
                              void* d_out, int out_size, void* d_ws, size_t ws_size,
                              hipStream_t stream) {
  (void)in_sizes; (void)n_in; (void)out_size; (void)ws_size;
  const float* x    = (const float*)d_in[0];
  const float* xt   = (const float*)d_in[1];
  const float* Wq   = (const float*)d_in[2];
  const float* Wk   = (const float*)d_in[3];
  const float* Wv   = (const float*)d_in[4];
  const float* Wout = (const float*)d_in[5];
  const float* W1   = (const float*)d_in[6];
  const float* b1   = (const float*)d_in[7];
  const float* gamma= (const float*)d_in[8];
  const float* beta = (const float*)d_in[9];
  const float* W2   = (const float*)d_in[10];
  const float* b2   = (const float*)d_in[11];
  float* outp = (float*)d_out;
  int*   idxp = (int*)d_out + (size_t)BB * OO * MM;   // idx appended after out

  char* ws = (char*)d_ws;
  size_t off = 0;
  auto alloc = [&](size_t bytes) {
    off = (off + 255) & ~(size_t)255;
    char* p = ws + off;
    off += bytes;
    return p;
  };
  h16*   xcT   = (h16*)alloc((size_t)BB * NN * KPAD * 2);
  h16*   centT = (h16*)alloc((size_t)BB * MM * KPAD * 2);
  h16*   qT    = (h16*)alloc((size_t)BB * MM * OO * 2);
  h16*   kTb   = (h16*)alloc((size_t)BB * NN * OO * 2);
  h16*   vR    = (h16*)alloc((size_t)BB * OO * NN * 2);
  h16*   attT  = (h16*)alloc((size_t)BB * MM * OO * 2);
  h16*   att2T = (h16*)alloc((size_t)BB * MM * OO * 2);
  float* h32   = (float*)alloc((size_t)BB * MM * OO * 4);
  h16*   Wq16  = (h16*)alloc((size_t)OO * KPAD * 2);
  h16*   Wk16  = (h16*)alloc((size_t)OO * KPAD * 2);
  h16*   Wv16  = (h16*)alloc((size_t)OO * KPAD * 2);
  h16*   Wo16  = (h16*)alloc((size_t)OO * OO * 2);
  h16*   W116  = (h16*)alloc((size_t)OO * OO * 2);
  h16*   W216  = (h16*)alloc((size_t)OO * OO * 2);
  float* ssum  = (float*)alloc(256 * 4);     // sum[128] ++ sumsq[128]
  float* scaleA= (float*)alloc(128 * 4);
  float* shiftA= (float*)alloc(128 * 4);

  // weights -> f16 (Wq carries log2e/sqrt(O))
  k_convw<<<dim3((OO * KPAD) / 256), 256, 0, stream>>>(Wq, Wq16, CC, KPAD, QSCALE);
  k_convw<<<dim3((OO * KPAD) / 256), 256, 0, stream>>>(Wk, Wk16, CC, KPAD, 1.f);
  k_convw<<<dim3((OO * KPAD) / 256), 256, 0, stream>>>(Wv, Wv16, CC, KPAD, 1.f);
  k_convw<<<dim3((OO * OO) / 256), 256, 0, stream>>>(Wout, Wo16, OO, OO, 1.f);
  k_convw<<<dim3((OO * OO) / 256), 256, 0, stream>>>(W1, W116, OO, OO, 1.f);
  k_convw<<<dim3((OO * OO) / 256), 256, 0, stream>>>(W2, W216, OO, OO, 1.f);

  k_concat<<<dim3((BB * NN) / 256), 256, 0, stream>>>(x, xt, xcT);
  k_fps<<<dim3(BB), 1024, 0, stream>>>(xcT, idxp);
  k_gather<<<dim3((BB * MM * KPAD) / 256), 256, 0, stream>>>(xcT, idxp, centT);

  // projections (WMMA): qT [B,M,128], kT [B,N,128], v [B,128,N]
  k_gemm<KPAD, 0, false><<<dim3(MM / 16, 2, BB), 128, 0, stream>>>(Wq16, centT, nullptr, qT, MM);
  k_gemm<KPAD, 0, false><<<dim3(NN / 16, 2, BB), 128, 0, stream>>>(Wk16, xcT, nullptr, kTb, NN);
  k_gemm<KPAD, 1, false><<<dim3(NN / 16, 2, BB), 128, 0, stream>>>(Wv16, xcT, nullptr, vR, NN);

  // flash attention -> attT [B,M,128]  (TDM-staged k/v tiles)
  k_flash<<<dim3(MM / 64, BB), 128, 0, stream>>>(qT, kTb, vR, attT);

  // Wout, then W1+b1 (f32 out for BN), stats, fused BN+leaky+W2+b2
  k_gemm<OO, 0, false><<<dim3(MM / 16, 2, BB), 128, 0, stream>>>(Wo16, attT, nullptr, att2T, MM);
  k_gemm<OO, 2, true ><<<dim3(MM / 16, 2, BB), 128, 0, stream>>>(W116, att2T, b1, h32, MM);
  k_zero<<<1, 256, 0, stream>>>(ssum, 256);
  k_stats<<<dim3(BM / 128), 256, 0, stream>>>(h32, ssum, ssum + 128);
  k_bnfin<<<1, 128, 0, stream>>>(ssum, ssum + 128, gamma, beta, scaleA, shiftA);
  k_out<<<dim3(MM / 16, 2, BB), 128, 0, stream>>>(W216, h32, scaleA, shiftA, b2, outp);
}
